// window_sparse_attention_36455682408883
// MI455X (gfx1250) — matrix operand order repaired and checked by reading, not compiled, not run on hardware
//
#include <hip/hip_runtime.h>

#define NTOK 64
#define DIMC 256
#define HEADS 8
#define HD 32
#define KSEL 32
#define SCALE 0.17677669529663687f   // 32^-0.5

typedef __attribute__((ext_vector_type(16))) __bf16 v16bf;
typedef __attribute__((ext_vector_type(8)))  float  v8f;

union AB16 { v16bf v; uint4 q[2]; unsigned short s[16]; };

__device__ __forceinline__ unsigned short f2bf(float f) {
  union { float f; unsigned u; } c; c.f = f;
  unsigned u = c.u;
  u += 0x7FFFu + ((u >> 16) & 1u);   // round-to-nearest-even
  return (unsigned short)(u >> 16);
}

// ---- LDS layout (bytes) ----
#define XSTR 264   // bf16 row stride for XB/QS/KS/OB (256 + pad)
#define VSTR 72    // bf16 row stride for VT  (64 tokens + pad)
#define SSTR 68    // f32  row stride for S   (64 + pad)
#define OFF_XB 0
#define OFF_QS (OFF_XB + 64*XSTR*2)          // 33792
#define OFF_KS (OFF_QS + 64*XSTR*2)          // 67584
#define OFF_VT (OFF_KS + 64*XSTR*2)          // 101376
#define OFF_OB (OFF_VT + 256*VSTR*2)         // 138240
#define OFF_S  (OFF_OB + 64*XSTR*2)          // 172032 (S also stages fp32 x)
#define LDS_BYTES (OFF_S + HEADS*64*SSTR*4)  // 311296 (< 320KB)

// ---- workspace layout (bytes) ----
#define WS_WQ   0                   // 768*256 bf16 = 393216
#define WS_WP   393216              // 256*256 bf16 = 131072
#define WS_BIAS 524288              // 8*64*64 f32  = 131072

__global__ void wsattn_prep_kernel(const float* __restrict__ wqkv,
                                   const float* __restrict__ wproj,
                                   const float* __restrict__ tab,
                                   const int*   __restrict__ idx,
                                   unsigned short* __restrict__ wq,
                                   unsigned short* __restrict__ wp,
                                   float* __restrict__ bias) {
  int i = blockIdx.x * blockDim.x + threadIdx.x;
  if (i < 3 * DIMC * DIMC) wq[i] = f2bf(wqkv[i]);
  if (i < DIMC * DIMC)     wp[i] = f2bf(wproj[i]);
  if (i < HEADS * NTOK * NTOK) {
    int h = i >> 12;            // / 4096
    int rem = i & 4095;         // q*64 + k
    bias[i] = tab[idx[rem] * HEADS + h];
  }
}

__global__ __launch_bounds__(256)
void wsattn_main_kernel(const float* __restrict__ x,
                        const float* __restrict__ bqkv,
                        const float* __restrict__ bproj,
                        const unsigned short* __restrict__ wq,
                        const unsigned short* __restrict__ wp,
                        const float* __restrict__ biasg,
                        float* __restrict__ out) {
  extern __shared__ char smem[];
  unsigned short* XB = (unsigned short*)(smem + OFF_XB);
  unsigned short* QS = (unsigned short*)(smem + OFF_QS);
  unsigned short* KS = (unsigned short*)(smem + OFF_KS);
  unsigned short* VT = (unsigned short*)(smem + OFF_VT);
  unsigned short* OB = (unsigned short*)(smem + OFF_OB);
  float*          S  = (float*)(smem + OFF_S);

  const int b    = blockIdx.x;
  const int tid  = threadIdx.x;
  const int lane = tid & 31;
  const int w    = tid >> 5;          // wave id 0..7
  const int ln15 = lane & 15;
  const int hi16 = (lane & 16) ? 1 : 0;

  // ---- Step A0: async-copy x tile (fp32, 64KB) HBM -> LDS (staged in S region)
  // GLOBAL_LOAD_ASYNC_TO_LDS_B128, GV mode, tracked by ASYNCcnt (no VGPR data).
  {
    const float* xb = x + (size_t)b * (NTOK * DIMC);
    unsigned sbase = (unsigned)(size_t)(void*)S;   // LDS byte address (low 32 bits)
#pragma unroll
    for (int k = 0; k < 16; k++) {
      int i = tid + k * 256;                       // float4 chunk index, 4096 total
      unsigned lds_off = sbase + (unsigned)(i * 16);
      unsigned long long ga = (unsigned long long)(const void*)(xb + i * 4);
      asm volatile("global_load_async_to_lds_b128 %0, %1, off"
                   :: "v"(lds_off), "v"(ga) : "memory");
    }
  }

  // Warm weight lines into caches while the async DMA runs (global_prefetch_b8)
  __builtin_prefetch(wq + tid * 768, 0, 3);
  __builtin_prefetch(wp + tid * 256, 0, 3);

  asm volatile("s_wait_asynccnt 0x0" ::: "memory");
  __syncthreads();

  // ---- Step A1: staged fp32 x -> bf16 XB (LDS -> LDS) ----
  {
    const float* XF = S;   // linear [64][256] fp32
    for (int i = tid; i < (NTOK * DIMC) / 4; i += 256) {
      float4 f = ((const float4*)XF)[i];
      int e = i * 4;
      int tok = e >> 8, c = e & 255;
      unsigned short* d = XB + tok * XSTR + c;
      d[0] = f2bf(f.x); d[1] = f2bf(f.y); d[2] = f2bf(f.z); d[3] = f2bf(f.w);
    }
  }
  __syncthreads();

  // ---- Step B: QKV = x @ Wqkv^T + b  (64x768), bf16 WMMA, fp32 acc ----
  for (int t = w; t < 4 * 48; t += 8) {
    int mt = t & 3, nt = t >> 2;
    v8f c = {};
    const unsigned short* aptr = XB + (mt * 16 + ln15) * XSTR + (hi16 ? 8 : 0);
    int o = nt * 16 + ln15;
    const unsigned short* brow = wq + o * DIMC + (hi16 ? 8 : 0);
#pragma unroll
    for (int ks = 0; ks < 8; ks++) {
      AB16 a, bb;
      a.q[0]  = *(const uint4*)(aptr + ks * 32);
      a.q[1]  = *(const uint4*)(aptr + ks * 32 + 16);
      bb.q[0] = *(const uint4*)(brow + ks * 32);
      bb.q[1] = *(const uint4*)(brow + ks * 32 + 16);
      c = __builtin_amdgcn_wmma_f32_16x16x32_bf16(false, a.v, false, bb.v,
                                                  (short)0, c, false, false);
    }
    float bias = bqkv[o];
    // which/cc base derived from nt only -> provably wave-uniform scalar branch
    int which = (nt * 16) >> 8;        // 0=q 1=k 2=v
    int cc = ((nt * 16) & 255) + ln15; // h*32 + d
#pragma unroll
    for (int r = 0; r < 8; r++) {
      int m = mt * 16 + r + hi16 * 8;
      float val = c[r] + bias;
      if (which == 0)      QS[m * XSTR + cc] = f2bf(val * SCALE);
      else if (which == 1) KS[m * XSTR + cc] = f2bf(val);
      else                 VT[cc * VSTR + m] = f2bf(val);   // V transposed
    }
  }
  __syncthreads();

  // ---- Step C: S_h = q_h k_h^T + bias (64x64 fp32), wave h = head h ----
  const int h = w;
  float* SH = S + h * (NTOK * SSTR);
  const unsigned short* qh = QS + h * HD;
  const unsigned short* kh = KS + h * HD;
  const float* bg = biasg + h * (NTOK * NTOK);
  for (int t = 0; t < 16; t++) {
    int mt = t & 3, nt = t >> 2;
    AB16 a, bb;
    const unsigned short* ap = qh + (mt * 16 + ln15) * XSTR + (hi16 ? 8 : 0);
    a.q[0] = *(const uint4*)(ap);
    a.q[1] = *(const uint4*)(ap + 16);
    const unsigned short* bp = kh + (nt * 16 + ln15) * XSTR + (hi16 ? 8 : 0);
    bb.q[0] = *(const uint4*)(bp);
    bb.q[1] = *(const uint4*)(bp + 16);
    v8f c = {};
    c = __builtin_amdgcn_wmma_f32_16x16x32_bf16(false, a.v, false, bb.v,
                                                (short)0, c, false, false);
#pragma unroll
    for (int r = 0; r < 8; r++) {
      int m = mt * 16 + r + hi16 * 8;
      int n = nt * 16 + ln15;
      SH[m * SSTR + n] = c[r] + bg[m * NTOK + n];
    }
  }
  __syncthreads();

  // ---- Step D: top-32-of-64 sparse softmax per row (rank counting) ----
  for (int row = 0; row < NTOK; row++) {
    float vlo = SH[row * SSTR + lane];
    float vhi = SH[row * SSTR + lane + 32];
    float m = fmaxf(vlo, vhi);
#pragma unroll
    for (int off = 16; off > 0; off >>= 1)
      m = fmaxf(m, __shfl_xor(m, off, 32));
    int rlo = 0, rhi = 0;
    for (int j = 0; j < 64; j++) {
      float ej = __shfl(j < 32 ? vlo : vhi, j & 31, 32);
      rlo += (ej > vlo) ? 1 : 0;
      rhi += (ej > vhi) ? 1 : 0;
    }
    float elo = (rlo < KSEL) ? __expf(vlo - m) : 0.0f;
    float ehi = (rhi < KSEL) ? __expf(vhi - m) : 0.0f;
    float s = elo + ehi;
#pragma unroll
    for (int off = 16; off > 0; off >>= 1)
      s += __shfl_xor(s, off, 32);
    float inv = 1.0f / s;           // max is always kept -> s >= 1
    SH[row * SSTR + lane]      = elo * inv;
    SH[row * SSTR + lane + 32] = ehi * inv;
  }
  __syncthreads();

  // ---- Step E: out_h = P_h @ v_h (64x32), P fp32->bf16 on load ----
  const unsigned short* vth = VT + (h * HD) * VSTR;
  for (int t = 0; t < 8; t++) {
    int mt = t & 3, nt = t >> 2;    // nt: 0..1
    v8f c = {};
#pragma unroll
    for (int ks = 0; ks < 2; ks++) {
      const float* pr = SH + (mt * 16 + ln15) * SSTR + ks * 32 + (hi16 ? 8 : 0);
      float4 f0 = *(const float4*)(pr);
      float4 f1 = *(const float4*)(pr + 4);
      float4 f2 = *(const float4*)(pr + 16);
      float4 f3 = *(const float4*)(pr + 20);
      AB16 a;
      a.s[0]  = f2bf(f0.x); a.s[1]  = f2bf(f0.y); a.s[2]  = f2bf(f0.z); a.s[3]  = f2bf(f0.w);
      a.s[4]  = f2bf(f1.x); a.s[5]  = f2bf(f1.y); a.s[6]  = f2bf(f1.z); a.s[7]  = f2bf(f1.w);
      a.s[8]  = f2bf(f2.x); a.s[9]  = f2bf(f2.y); a.s[10] = f2bf(f2.z); a.s[11] = f2bf(f2.w);
      a.s[12] = f2bf(f3.x); a.s[13] = f2bf(f3.y); a.s[14] = f2bf(f3.z); a.s[15] = f2bf(f3.w);
      const unsigned short* bp = vth + (nt * 16 + ln15) * VSTR + ks * 32 + (hi16 ? 8 : 0);
      AB16 bb;
      bb.q[0] = *(const uint4*)bp;
      bb.q[1] = *(const uint4*)(bp + 16);
      c = __builtin_amdgcn_wmma_f32_16x16x32_bf16(false, a.v, false, bb.v,
                                                  (short)0, c, false, false);
    }
#pragma unroll
    for (int r = 0; r < 8; r++) {
      int m = mt * 16 + r + hi16 * 8;
      int d = h * HD + nt * 16 + ln15;
      OB[m * XSTR + d] = f2bf(c[r]);
    }
  }
  __syncthreads();

  // ---- Step F: Y = OB @ Wproj^T + b, stream fp32 to HBM ----
  float* orow = out + (size_t)b * (NTOK * DIMC);
  for (int t = w; t < 64; t += 8) {
    int mt = t & 3, nt = t >> 2;
    v8f c = {};
    int o = nt * 16 + ln15;
    const unsigned short* ap0 = OB + (mt * 16 + ln15) * XSTR + (hi16 ? 8 : 0);
    const unsigned short* bp0 = wp + o * DIMC + (hi16 ? 8 : 0);
#pragma unroll
    for (int ks = 0; ks < 8; ks++) {
      AB16 a, bb;
      a.q[0]  = *(const uint4*)(ap0 + ks * 32);
      a.q[1]  = *(const uint4*)(ap0 + ks * 32 + 16);
      bb.q[0] = *(const uint4*)(bp0 + ks * 32);
      bb.q[1] = *(const uint4*)(bp0 + ks * 32 + 16);
      c = __builtin_amdgcn_wmma_f32_16x16x32_bf16(false, a.v, false, bb.v,
                                                  (short)0, c, false, false);
    }
    float bias = bproj[o];
#pragma unroll
    for (int r = 0; r < 8; r++) {
      int m = mt * 16 + r + hi16 * 8;
      orow[m * DIMC + o] = c[r] + bias;
    }
  }
}

extern "C" void kernel_launch(void* const* d_in, const int* in_sizes, int n_in,
                              void* d_out, int out_size, void* d_ws, size_t ws_size,
                              hipStream_t stream) {
  const float* x     = (const float*)d_in[0];
  const float* tab   = (const float*)d_in[1];
  const float* wqkv  = (const float*)d_in[2];
  const float* bqkv  = (const float*)d_in[3];
  const float* wproj = (const float*)d_in[4];
  const float* bproj = (const float*)d_in[5];
  const int*   idx   = (const int*)d_in[6];

  unsigned short* wq   = (unsigned short*)((char*)d_ws + WS_WQ);
  unsigned short* wp   = (unsigned short*)((char*)d_ws + WS_WP);
  float*          bias = (float*)((char*)d_ws + WS_BIAS);

  wsattn_prep_kernel<<<768, 256, 0, stream>>>(wqkv, wproj, tab, idx, wq, wp, bias);

  (void)hipFuncSetAttribute(reinterpret_cast<const void*>(&wsattn_main_kernel),
                            hipFuncAttributeMaxDynamicSharedMemorySize, LDS_BYTES);
  wsattn_main_kernel<<<2048, 256, LDS_BYTES, stream>>>(x, bqkv, bproj, wq, wp, bias,
                                                       (float*)d_out);
}
